// KNN_InstanceLoss_51247549776209
// MI455X (gfx1250) — matrix operand order repaired and checked by reading, not compiled, not run on hardware
//
#include <hip/hip_runtime.h>
#include <hip/hip_bf16.h>

typedef __bf16 bf16_t;
typedef __attribute__((ext_vector_type(16))) __bf16 v16bf;
typedef __attribute__((ext_vector_type(8)))  __bf16 v8bf;
typedef __attribute__((ext_vector_type(8)))  float  v8f;

#define BB 4096        // batch
#define DD 256         // embedding dim
#define CC 128         // cluster dim
#define NSLICE 4       // column slices (grid.y)
#define RPB 64         // rows per block (4 waves x 16)
#define CAPS 8         // stored positives per row per slice
#define INV_T 2.0f     // 1 / TEMPERATURE
#define THRESH 0.5f

// ---------------------------------------------------------------------------
// Kernel 1: fp32 -> bf16 conversion into workspace
// ---------------------------------------------------------------------------
__global__ void cvt_bf16_kernel(const float* __restrict__ src,
                                bf16_t* __restrict__ dst, int n) {
  int i = blockIdx.x * blockDim.x + threadIdx.x;
  if (i < n) dst[i] = (bf16_t)src[i];
}

// ---------------------------------------------------------------------------
// Kernel 2: fused mask-GEMM + cos-GEMM + masked exp-sum.
// Grid: (64 row-blocks, 4 column-slices). Block = 128 threads = 4 waves.
// Wave w owns row band [blockIdx.x*64 + 16w, +16); all 4 waves sweep the SAME
// 64 column tiles of their slice -> B operand loads are shared via WGP caches
// (4x less L2 traffic than one-band-per-block). Per-row partial results are
// written per-slice and merged in fixed order by the finalize kernel.
// ---------------------------------------------------------------------------
__global__ __launch_bounds__(128) void band_kernel(
    const bf16_t* __restrict__ zi, const bf16_t* __restrict__ zj,
    const bf16_t* __restrict__ ci, const bf16_t* __restrict__ cj,
    float* __restrict__ S_part, int* __restrict__ cnt_part,
    float* __restrict__ pos_part)
{
  __shared__ int   sh_cnt[RPB];
  __shared__ float sh_pos[RPB * CAPS];

  const int tid   = threadIdx.x;
  const int wave  = tid >> 5;
  const int lane  = tid & 31;
  const int hi    = (lane >> 4) & 1;  // 16-lane half
  const int ln    = lane & 15;        // row (A) / col (B,C) index within tile
  const int slice = blockIdx.y;
  const int m0    = blockIdx.x * RPB + wave * 16;   // this wave's row band
  const int nt0   = slice * (BB / 16 / NSLICE);     // first column tile
  const int ntN   = nt0 + (BB / 16 / NSLICE);       // one past last

  if (tid < RPB) sh_cnt[tid] = 0;
  __syncthreads();

  // ---- Preload A fragments for this wave's band (kept in VGPRs) ----
  // A 16x32 bf16 layout: lane(hi,m) holds k in [16hi,16hi+16) contiguously.
  v16bf aZ[8], aCi[4], aCj[4];
  {
    const bf16_t* rz = zi + (size_t)(m0 + ln) * DD;
#pragma unroll
    for (int kb = 0; kb < 8; ++kb) {
      const bf16_t* p = rz + kb * 32 + 16 * hi;
      v8bf lo = *(const v8bf*)p;
      v8bf hh = *(const v8bf*)(p + 8);
#pragma unroll
      for (int i = 0; i < 8; ++i) { aZ[kb][i] = lo[i]; aZ[kb][8 + i] = hh[i]; }
    }
    const bf16_t* rci = ci + (size_t)(m0 + ln) * CC;
    const bf16_t* rcj = cj + (size_t)(m0 + ln) * CC;
#pragma unroll
    for (int kb = 0; kb < 4; ++kb) {
      const bf16_t* p = rci + kb * 32 + 16 * hi;
      v8bf lo = *(const v8bf*)p;
      v8bf hh = *(const v8bf*)(p + 8);
#pragma unroll
      for (int i = 0; i < 8; ++i) { aCi[kb][i] = lo[i]; aCi[kb][8 + i] = hh[i]; }
      const bf16_t* q = rcj + kb * 32 + 16 * hi;
      v8bf lo2 = *(const v8bf*)q;
      v8bf hh2 = *(const v8bf*)(q + 8);
#pragma unroll
      for (int i = 0; i < 8; ++i) { aCj[kb][i] = lo2[i]; aCj[kb][8 + i] = hh2[i]; }
    }
  }

  float Sl[8];
#pragma unroll
  for (int r = 0; r < 8; ++r) Sl[r] = 0.0f;

  // ---- Column-tile sweep (identical sequence for all 4 waves) ----
  for (int nt = nt0; nt < ntN; ++nt) {
    const int n0 = nt * 16;
    // B 32x16 bf16 layout: lane(hi,n) holds k in [16hi,16hi+16) contiguously.
    const bf16_t* colz  = zj + (size_t)(n0 + ln) * DD + 16 * hi;
    const bf16_t* colci = ci + (size_t)(n0 + ln) * CC + 16 * hi;
    const bf16_t* colcj = cj + (size_t)(n0 + ln) * CC + 16 * hi;

    // label mask tile: c_i c_i^T + c_j c_j^T  (K = 128 each -> 4+4 WMMAs)
    v8f mk = {};
#pragma unroll
    for (int kb = 0; kb < 4; ++kb) {
      v16bf b = *(const v16bf*)(colci + kb * 32);
      mk = __builtin_amdgcn_wmma_f32_16x16x32_bf16(false, aCi[kb], false, b,
                                                   (short)0, mk, false, false);
    }
#pragma unroll
    for (int kb = 0; kb < 4; ++kb) {
      v16bf b = *(const v16bf*)(colcj + kb * 32);
      mk = __builtin_amdgcn_wmma_f32_16x16x32_bf16(false, aCj[kb], false, b,
                                                   (short)0, mk, false, false);
    }
    // cosine tile: z_i z_j^T  (K = 256 -> 8 WMMAs)
    v8f cs = {};
#pragma unroll
    for (int kb = 0; kb < 8; ++kb) {
      v16bf b = *(const v16bf*)(colz + kb * 32);
      cs = __builtin_amdgcn_wmma_f32_16x16x32_bf16(false, aZ[kb], false, b,
                                                   (short)0, cs, false, false);
    }

    // C/D layout: lane l, vgpr r -> element (M = r + 8*(l>>4), N = l&15)
    const int col = n0 + ln;
#pragma unroll
    for (int r = 0; r < 8; ++r) {
      const int row = m0 + r + 8 * hi;
      const float c = cs[r];
      const bool pos = (row == col) || (0.5f * mk[r] > THRESH);
      if (pos) {
        const int rb = wave * 16 + r + 8 * hi;       // row within block
        int slot = atomicAdd(&sh_cnt[rb], 1);
        if (slot < CAPS) sh_pos[rb * CAPS + slot] = c;
      } else {
        Sl[r] += __expf(c * INV_T);
      }
    }
  }

  // ---- Deterministic xor-tree reduction over the 16 lanes of each half ----
#pragma unroll
  for (int r = 0; r < 8; ++r) {
    float v = Sl[r];
    v += __shfl_xor(v, 1, 32);
    v += __shfl_xor(v, 2, 32);
    v += __shfl_xor(v, 4, 32);
    v += __shfl_xor(v, 8, 32);
    Sl[r] = v;
  }
  if (ln == 0) {   // lane 0 -> rows m0+0..7, lane 16 -> rows m0+8..15
#pragma unroll
    for (int r = 0; r < 8; ++r)
      S_part[(size_t)slice * BB + m0 + r + 8 * hi] = Sl[r];
  }
  __syncthreads();

  const int base = slice * BB + blockIdx.x * RPB;    // per-slice row index
  if (tid < RPB) cnt_part[base + tid] = sh_cnt[tid];
  for (int i = tid; i < RPB * CAPS; i += 128) {
    int rr = i / CAPS, sl = i % CAPS;
    int avail = min(sh_cnt[rr], CAPS);
    pos_part[(size_t)(base + rr) * CAPS + sl] = (sl < avail) ? sh_pos[i] : -3.0e38f;
  }
}

// ---------------------------------------------------------------------------
// Kernel 3: merge slices (fixed order), pos_min, per-row top-k, final loss.
// Single block, fixed-order tree reductions -> deterministic.
// ---------------------------------------------------------------------------
__global__ __launch_bounds__(256) void finalize_kernel(
    const float* __restrict__ S_part, const int* __restrict__ cnt_part,
    const float* __restrict__ pos_part, float* __restrict__ out)
{
  __shared__ int   smin[256];
  __shared__ float ssum[256];
  __shared__ int   s_posmin;

  const int tid = threadIdx.x;

  // pass 1: pos_min over merged per-row counts
  int lmin = 0x7fffffff;
  for (int b = tid; b < BB; b += 256) {
    int c = 0;
#pragma unroll
    for (int s = 0; s < NSLICE; ++s) c += cnt_part[(size_t)s * BB + b];
    lmin = min(lmin, c);
  }
  smin[tid] = lmin;
  __syncthreads();
  for (int s = 128; s > 0; s >>= 1) {
    if (tid < s) smin[tid] = min(smin[tid], smin[tid + s]);
    __syncthreads();
  }
  if (tid == 0) s_posmin = max(smin[0], 1);
  __syncthreads();
  const int pmin = s_posmin;

  // pass 2: per-row loss terms
  float local = 0.0f;
  for (int b = tid; b < BB; b += 256) {
    float Sb = 0.0f;
    float vals[NSLICE * CAPS];
    int avail = 0;
#pragma unroll
    for (int s = 0; s < NSLICE; ++s) {
      Sb += S_part[(size_t)s * BB + b];                  // fixed slice order
      const int c = min(cnt_part[(size_t)s * BB + b], CAPS);
      for (int i = 0; i < c; ++i)
        vals[avail++] = pos_part[((size_t)s * BB + b) * CAPS + i];
    }
    const int kk = min(pmin, avail);
    for (int p = 0; p < kk; ++p) {           // partial selection sort (desc)
      int best = p;
      for (int q = p + 1; q < avail; ++q)
        if (vals[q] > vals[best]) best = q;
      float t = vals[p]; vals[p] = vals[best]; vals[best] = t;
      const float v2 = vals[p] * INV_T;
      local += logf(__expf(v2) + Sb) - v2;   // logsumexp - logits[0]
    }
  }
  ssum[tid] = local;
  __syncthreads();
  for (int s = 128; s > 0; s >>= 1) {
    if (tid < s) ssum[tid] += ssum[tid + s];
    __syncthreads();
  }
  if (tid == 0) out[0] = ssum[0] / (float)((size_t)BB * (size_t)pmin);
}

// ---------------------------------------------------------------------------
extern "C" void kernel_launch(void* const* d_in, const int* in_sizes, int n_in,
                              void* d_out, int out_size, void* d_ws, size_t ws_size,
                              hipStream_t stream) {
  (void)in_sizes; (void)n_in; (void)out_size; (void)ws_size;
  const float* zi = (const float*)d_in[0];
  const float* zj = (const float*)d_in[1];
  const float* ci = (const float*)d_in[2];
  const float* cj = (const float*)d_in[3];

  char* ws = (char*)d_ws;
  size_t off = 0;
  bf16_t* zib = (bf16_t*)(ws + off); off += (size_t)BB * DD * sizeof(bf16_t);
  bf16_t* zjb = (bf16_t*)(ws + off); off += (size_t)BB * DD * sizeof(bf16_t);
  bf16_t* cib = (bf16_t*)(ws + off); off += (size_t)BB * CC * sizeof(bf16_t);
  bf16_t* cjb = (bf16_t*)(ws + off); off += (size_t)BB * CC * sizeof(bf16_t);
  float*  Sp  = (float*)(ws + off);  off += (size_t)NSLICE * BB * sizeof(float);
  int*    cp  = (int*)(ws + off);    off += (size_t)NSLICE * BB * sizeof(int);
  float*  pp  = (float*)(ws + off);  off += (size_t)NSLICE * BB * CAPS * sizeof(float);

  const int nz = BB * DD, nc = BB * CC;
  cvt_bf16_kernel<<<(nz + 255) / 256, 256, 0, stream>>>(zi, zib, nz);
  cvt_bf16_kernel<<<(nz + 255) / 256, 256, 0, stream>>>(zj, zjb, nz);
  cvt_bf16_kernel<<<(nc + 255) / 256, 256, 0, stream>>>(ci, cib, nc);
  cvt_bf16_kernel<<<(nc + 255) / 256, 256, 0, stream>>>(cj, cjb, nc);

  band_kernel<<<dim3(BB / RPB, NSLICE), 128, 0, stream>>>(zib, zjb, cib, cjb,
                                                          Sp, cp, pp);
  finalize_kernel<<<1, 256, 0, stream>>>(Sp, cp, pp, (float*)d_out);
}
